// TransitionPathNetwork_66675072303210
// MI455X (gfx1250) — matrix operand order repaired and checked by reading, not compiled, not run on hardware
//
#include <hip/hip_runtime.h>
#include <hip/hip_bf16.h>
#include <math.h>

// ---------------------------------------------------------------------------
// TransitionPathNetwork forward for MI455X (gfx1250, wave32, WMMA 16x16x32 f16)
// Pair MLP first layer decomposed into node projections U,V + edge term.
// Pair kernel fuses: h1 = gelu(U_i + V_j + e.We) -> WMMA 64x64 -> gelu
//                    -> WMMA 64x64 -> masked j-reduction (LDS atomics).
// Branch-free GELU: x * rcp(1+exp(-u)) with native v_exp_f32 + v_rcp_f32.
// Weights pre-swizzled in LDS to per-lane fragment order (2x ds_load_b128).
// ---------------------------------------------------------------------------

#define NN    512
#define STATE 208

typedef __attribute__((ext_vector_type(16))) _Float16 v16h;
typedef __attribute__((ext_vector_type(8)))  _Float16 h8;
typedef __attribute__((ext_vector_type(8)))  float    v8f;
typedef __attribute__((ext_vector_type(4)))  float    f32x4;

// gelu(tanh approx) == x * sigmoid(2*0.7978845608*(x + 0.044715 x^3))
// Native exp + native reciprocal: no libm tanh, no IEEE div sequence.
__device__ __forceinline__ float gelu_f(float x) {
  float u = x * (1.5957691216057308f + 0.07135481627f * x * x);
  return x * __builtin_amdgcn_rcpf(1.0f + __expf(-u));
}

// Build one A-fragment (16x32 f16, documented layout) of h1 values:
// h1[k] = gelu(U[k] + V[k] + sum_f e[f]*E[f][k]); per lane K's are two
// contiguous runs of 8 -> float4 loads.
__device__ __forceinline__ v16h h1_frag(const float* __restrict__ U,
                                        const float* __restrict__ V,
                                        const float* __restrict__ E,
                                        const float e[6], int kbase, int khalf) {
  v16h a;
#pragma unroll
  for (int r = 0; r < 2; ++r) {
    const int base = kbase + 16 * r + khalf;
    const f32x4* U4 = (const f32x4*)(U + base);
    const f32x4* V4 = (const f32x4*)(V + base);
    f32x4 v0 = U4[0] + V4[0];
    f32x4 v1 = U4[1] + V4[1];
#pragma unroll
    for (int f = 0; f < 6; ++f) {
      const f32x4* E4 = (const f32x4*)(E + f * 64 + base);
      v0 += E4[0] * e[f];
      v1 += E4[1] * e[f];
    }
#pragma unroll
    for (int q = 0; q < 4; ++q) {
      a[8 * r + q]     = (_Float16)gelu_f(v0[q]);
      a[8 * r + 4 + q] = (_Float16)gelu_f(v1[q]);
    }
  }
  return a;
}

// B fragment from pre-swizzled LDS (per-lane 16 halves contiguous, 32B aligned)
__device__ __forceinline__ v16h load_bf(const _Float16* __restrict__ Wf, int tile, int lane) {
  const h8* p = (const h8*)(Wf + ((tile * 32 + lane) << 4));
  h8 lo = p[0], hi = p[1];
  return __builtin_shufflevector(lo, hi, 0, 1, 2, 3, 4, 5, 6, 7, 8, 9, 10, 11, 12, 13, 14, 15);
}

// A fragment from a 16x64 row-major f16 tile in LDS: two 16B contiguous runs.
__device__ __forceinline__ v16h load_af(const _Float16* __restrict__ H, int kbase, int lane) {
  const int m = lane & 15, khalf = (lane >> 4) * 8;
  const h8* p0 = (const h8*)(H + m * 64 + kbase + khalf);
  const h8* p1 = (const h8*)(H + m * 64 + kbase + 16 + khalf);
  h8 lo = p0[0], hi = p1[0];
  return __builtin_shufflevector(lo, hi, 0, 1, 2, 3, 4, 5, 6, 7, 8, 9, 10, 11, 12, 13, 14, 15);
}

// ---------------------------------------------------------------------------
// Kernel 1: node embeddings -> h[N,208], base, x
// ---------------------------------------------------------------------------
struct EmbArgs {
  const int* Z; const float* xA; const float* xB; const float* s;
  const float* table;
  const float* Wi0; const float* bi0; const float* Wi1; const float* bi1;
  const float* WA0; const float* bA0; const float* WA1; const float* bA1;
  const float* WB0; const float* bB0; const float* WB1; const float* bB1;
  float* h; float* base; float* x;
};

__global__ __launch_bounds__(64) void node_embed_kernel(EmbArgs A) {
  const int i = blockIdx.x, t = threadIdx.x;
  __shared__ float zin[32];
  __shared__ float hid[64];
  if (t < 32) zin[t] = A.table[A.Z[i] * 32 + t];
  __syncthreads();
  const float sc = A.s[0];

  float acc = A.bi0[t];
  for (int c = 0; c < 32; ++c) acc += zin[c] * A.Wi0[c * 64 + t];
  hid[t] = gelu_f(acc);
  __syncthreads();
  float emb = A.bi1[t];
  for (int c = 0; c < 64; ++c) emb += hid[c] * A.Wi1[c * 64 + t];
  __syncthreads();

  float pA[3] = { A.xA[i * 3 + 0], A.xA[i * 3 + 1], A.xA[i * 3 + 2] };
  acc = A.bA0[t];
  for (int c = 0; c < 32; ++c) acc += zin[c] * A.WA0[c * 64 + t];
  for (int c = 0; c < 3; ++c)  acc += pA[c] * A.WA0[(32 + c) * 64 + t];
  hid[t] = gelu_f(acc);
  __syncthreads();
  float hA = A.bA1[t];
  for (int c = 0; c < 64; ++c) hA += hid[c] * A.WA1[c * 64 + t];
  __syncthreads();

  float pB[3] = { A.xB[i * 3 + 0], A.xB[i * 3 + 1], A.xB[i * 3 + 2] };
  acc = A.bB0[t];
  for (int c = 0; c < 32; ++c) acc += zin[c] * A.WB0[c * 64 + t];
  for (int c = 0; c < 3; ++c)  acc += pB[c] * A.WB0[(32 + c) * 64 + t];
  hid[t] = gelu_f(acc);
  __syncthreads();
  float hB = A.bB1[t];
  for (int c = 0; c < 64; ++c) hB += hid[c] * A.WB1[c * 64 + t];

  A.h[i * STATE + t]       = emb;
  A.h[i * STATE + 64 + t]  = hA;
  A.h[i * STATE + 128 + t] = hB;
  if (t < 16) {
    float k   = (float)((t & 7) + 1);
    float ang = 6.283185307179586f * k * sc;
    A.h[i * STATE + 192 + t] = (t < 8) ? sinf(ang) : cosf(ang);
  }
  if (t < 3) {
    float b = (1.0f - sc) * pA[t] + sc * pB[t];
    A.base[i * 3 + t] = b;
    A.x[i * 3 + t]    = b;
  }
}

// ---------------------------------------------------------------------------
// Kernel 2 (per layer): node projections (layer-0 bias folded into U) + beta
// ---------------------------------------------------------------------------
struct ProjArgs {
  const float* h;
  const float* W0m; const float* b0m;   // msg   layer0 [422,64] + bias
  const float* W0a; const float* b0a;   // alpha layer0
  const float* Wb0; const float* bb0; const float* Wb1; const float* bb1;
  const float* Wb2; const float* bb2;
  float* Um; float* Vm; float* Ua; float* Va; float* beta;
};

__global__ __launch_bounds__(64) void node_proj_kernel(ProjArgs A) {
  const int i = blockIdx.x, t = threadIdx.x;
  __shared__ float hr[STATE];
  __shared__ float hid1[64], hid2[64];
  for (int c = t; c < STATE; c += 64) hr[c] = A.h[i * STATE + c];
  __syncthreads();

  float um = A.b0m[t], vm = 0.f, ua = A.b0a[t], va = 0.f;
  for (int c = 0; c < STATE; ++c) {
    float hv = hr[c];
    um += hv * A.W0m[c * 64 + t];
    vm += hv * A.W0m[(STATE + c) * 64 + t];
    ua += hv * A.W0a[c * 64 + t];
    va += hv * A.W0a[(STATE + c) * 64 + t];
  }
  A.Um[i * 64 + t] = um; A.Vm[i * 64 + t] = vm;
  A.Ua[i * 64 + t] = ua; A.Va[i * 64 + t] = va;

  float acc = A.bb0[t];
  for (int c = 0; c < STATE; ++c) acc += hr[c] * A.Wb0[c * 64 + t];
  hid1[t] = gelu_f(acc);
  __syncthreads();
  acc = A.bb1[t];
  for (int c = 0; c < 64; ++c) acc += hid1[c] * A.Wb1[c * 64 + t];
  hid2[t] = gelu_f(acc);
  __syncthreads();
  if (t == 0) {
    float b = A.bb2[0];
    for (int c = 0; c < 64; ++c) b += hid2[c] * A.Wb2[c];
    A.beta[i] = b;
  }
}

// ---------------------------------------------------------------------------
// Kernel 3 (per layer): fused pair MLPs with WMMA; outputs m[N,64], dx[N,3]
// ---------------------------------------------------------------------------
struct PairArgs {
  const float* x; const float* xA; const float* xB;
  const unsigned char* bmA; const unsigned char* bmB;
  const float* Um; const float* Vm; const float* Ua; const float* Va;
  const float* W0m; const float* W2m; const float* b2m;
  const float* W3m; const float* b3m;
  const float* W0a; const float* W2a; const float* b2a;
  const float* w3a; const float* b3a;
  float* mOut; float* dxOut;
};

__global__ __launch_bounds__(128) void pair_kernel(PairArgs A) {
  const int i    = blockIdx.x;
  const int tid  = threadIdx.x;
  const int lane = tid & 31;
  const int w    = tid >> 5;

  // weights pre-swizzled to per-lane fragment order: [tile(kt*4+nt)][lane][16]
  __shared__ __align__(32) _Float16 sW2mF[4096], sW3mF[4096], sW2aF[4096];
  __shared__ __align__(16) float sEm[6 * 64], sEa[6 * 64];
  __shared__ __align__(16) float sB2m[64], sB3m[64], sB2a[64], sW3a[64];
  __shared__ __align__(16) float sUm[64], sUa[64];
  __shared__ float sXi[3], sXAi[3], sXBi[3];
  __shared__ float sMacc[64], sDx[3], sB3a;
  __shared__ __align__(32) _Float16 sH2[4][16 * 64];
  __shared__ float sAcc[4][16];
  __shared__ float sMask[4][16];
  __shared__ float sDir[4][48];

  for (int idx = tid; idx < 4096; idx += 128) {
    const int e_ = idx & 15;
    const int ln = (idx >> 4) & 31;
    const int tl = idx >> 9;                    // 0..7 = kt*4 + nt
    const int n  = (tl & 3) * 16 + (ln & 15);
    const int k  = (tl >> 2) * 32 + ((e_ >> 3) << 4) + ((ln >> 4) << 3) + (e_ & 7);
    const int src = k * 64 + n;
    sW2mF[idx] = (_Float16)A.W2m[src];
    sW3mF[idx] = (_Float16)A.W3m[src];
    sW2aF[idx] = (_Float16)A.W2a[src];
  }
  for (int t = tid; t < 6 * 64; t += 128) {
    const int f = t >> 6, k = t & 63;
    sEm[t] = A.W0m[(416 + f) * 64 + k];
    sEa[t] = A.W0a[(416 + f) * 64 + k];
  }
  if (tid < 64) {
    sB2m[tid] = A.b2m[tid]; sB3m[tid] = A.b3m[tid];
    sB2a[tid] = A.b2a[tid]; sW3a[tid] = A.w3a[tid];
    sUm[tid]  = A.Um[i * 64 + tid];
    sUa[tid]  = A.Ua[i * 64 + tid];
    sMacc[tid] = 0.f;
  }
  if (tid < 3) {
    sXi[tid] = A.x[i * 3 + tid]; sXAi[tid] = A.xA[i * 3 + tid]; sXBi[tid] = A.xB[i * 3 + tid];
    sDx[tid] = 0.f;
  }
  if (tid == 3) sB3a = A.b3a[0];
  __syncthreads();

  const int mloc  = lane & 15;
  const int khalf = (lane >> 4) * 8;
  const int moff  = (lane >> 4) << 3;

  for (int it = 0; it < 8; ++it) {
    const int j0 = (it * 4 + w) * 16;
    const int j  = j0 + mloc;

    // edge features for pair (i, j) — native v_sqrt_f32
    float xj0 = A.x[j * 3 + 0], xj1 = A.x[j * 3 + 1], xj2 = A.x[j * 3 + 2];
    float d0 = xj0 - sXi[0], d1 = xj1 - sXi[1], d2 = xj2 - sXi[2];
    float dsq = d0 * d0 + d1 * d1 + d2 * d2;
    float d   = (i == j) ? 0.f : __builtin_amdgcn_sqrtf(dsq);
    float a0 = A.xA[j * 3 + 0] - sXAi[0], a1 = A.xA[j * 3 + 1] - sXAi[1], a2 = A.xA[j * 3 + 2] - sXAi[2];
    float dAv = (i == j) ? 0.f : __builtin_amdgcn_sqrtf(a0 * a0 + a1 * a1 + a2 * a2);
    float c0 = A.xB[j * 3 + 0] - sXBi[0], c1 = A.xB[j * 3 + 1] - sXBi[1], c2 = A.xB[j * 3 + 2] - sXBi[2];
    float dBv = (i == j) ? 0.f : __builtin_amdgcn_sqrtf(c0 * c0 + c1 * c1 + c2 * c2);
    float bA = A.bmA[i * NN + j] ? 1.f : 0.f;
    float bB = A.bmB[i * NN + j] ? 1.f : 0.f;
    float e[6] = { bA, bB, d, d * d, dAv, dBv };
    float mask = (bA != 0.f || bB != 0.f || (d <= 2.0f && i != j)) ? 1.f : 0.f;
    if (lane < 16) {
      sMask[w][mloc] = mask;
      sDir[w][mloc * 3 + 0] = d0; sDir[w][mloc * 3 + 1] = d1; sDir[w][mloc * 3 + 2] = d2;
      sAcc[w][mloc] = 0.f;
    }
    __syncthreads();

    // h1 (layer 1) built directly in WMMA A-fragment layout
    const float* Vmj = A.Vm + j * 64;
    const float* Vaj = A.Va + j * 64;
    v16h am0 = h1_frag(sUm, Vmj, sEm, e, 0,  khalf);
    v16h am1 = h1_frag(sUm, Vmj, sEm, e, 32, khalf);
    v16h aa0 = h1_frag(sUa, Vaj, sEa, e, 0,  khalf);
    v16h aa1 = h1_frag(sUa, Vaj, sEa, e, 32, khalf);

    // layer 2 (64->64) msg + alpha via WMMA; alpha dotted with w3a on the fly
    float pa[8] = { 0.f, 0.f, 0.f, 0.f, 0.f, 0.f, 0.f, 0.f };
#pragma unroll
    for (int n0t = 0; n0t < 4; ++n0t) {
      const int n = n0t * 16 + mloc;
      {
        v16h b0 = load_bf(sW2mF, n0t,     lane);
        v16h b1 = load_bf(sW2mF, 4 + n0t, lane);
        v8f c = { 0.f, 0.f, 0.f, 0.f, 0.f, 0.f, 0.f, 0.f };
        c = __builtin_amdgcn_wmma_f32_16x16x32_f16(false, am0, false, b0, (short)0, c, false, false);
        c = __builtin_amdgcn_wmma_f32_16x16x32_f16(false, am1, false, b1, (short)0, c, false, false);
#pragma unroll
        for (int v = 0; v < 8; ++v)
          sH2[w][(v + moff) * 64 + n] = (_Float16)gelu_f(c[v] + sB2m[n]);
      }
      {
        v16h b0 = load_bf(sW2aF, n0t,     lane);
        v16h b1 = load_bf(sW2aF, 4 + n0t, lane);
        v8f c = { 0.f, 0.f, 0.f, 0.f, 0.f, 0.f, 0.f, 0.f };
        c = __builtin_amdgcn_wmma_f32_16x16x32_f16(false, aa0, false, b0, (short)0, c, false, false);
        c = __builtin_amdgcn_wmma_f32_16x16x32_f16(false, aa1, false, b1, (short)0, c, false, false);
        const float w3 = sW3a[n];
#pragma unroll
        for (int v = 0; v < 8; ++v)
          pa[v] += gelu_f(c[v] + sB2a[n]) * w3;
      }
    }
#pragma unroll
    for (int v = 0; v < 8; ++v) atomicAdd(&sAcc[w][v + moff], pa[v]);
    __syncthreads();

    // msg layer 3 (64->64) + masked reduction over the 16 j's
    v16h h0 = load_af(sH2[w], 0,  lane);
    v16h h1 = load_af(sH2[w], 32, lane);
#pragma unroll
    for (int n0t = 0; n0t < 4; ++n0t) {
      const int n = n0t * 16 + mloc;
      v16h b0 = load_bf(sW3mF, n0t,     lane);
      v16h b1 = load_bf(sW3mF, 4 + n0t, lane);
      v8f c = { 0.f, 0.f, 0.f, 0.f, 0.f, 0.f, 0.f, 0.f };
      c = __builtin_amdgcn_wmma_f32_16x16x32_f16(false, h0, false, b0, (short)0, c, false, false);
      c = __builtin_amdgcn_wmma_f32_16x16x32_f16(false, h1, false, b1, (short)0, c, false, false);
      float local = 0.f;
#pragma unroll
      for (int v = 0; v < 8; ++v)
        local += sMask[w][v + moff] * (c[v] + sB3m[n]);
      atomicAdd(&sMacc[n], local);
    }
    if (lane < 16) {
      float aval = (sAcc[w][mloc] + sB3a) * sMask[w][mloc];
      atomicAdd(&sDx[0], aval * sDir[w][mloc * 3 + 0]);
      atomicAdd(&sDx[1], aval * sDir[w][mloc * 3 + 1]);
      atomicAdd(&sDx[2], aval * sDir[w][mloc * 3 + 2]);
    }
    __syncthreads();
  }

  if (tid < 64) A.mOut[i * 64 + tid] = sMacc[tid];
  if (tid < 3)  A.dxOut[i * 3 + tid] = sDx[tid];
}

// ---------------------------------------------------------------------------
// Kernel 4 (per layer): x += dx + beta*(base-x); h += stateMLP([h, m])
// ---------------------------------------------------------------------------
struct UpdArgs {
  const float* m; const float* dx; const float* beta; const float* base;
  const float* Ws0; const float* bs0; const float* Ws1; const float* bs1;
  const float* Ws2; const float* bs2;
  float* h; float* x;
};

__global__ __launch_bounds__(64) void update_kernel(UpdArgs A) {
  const int i = blockIdx.x, t = threadIdx.x;
  __shared__ float sIn[272];
  __shared__ float hid1[64], hid2[64];
  for (int c = t; c < STATE; c += 64) sIn[c] = A.h[i * STATE + c];
  sIn[STATE + t] = A.m[i * 64 + t];
  __syncthreads();

  float acc = A.bs0[t];
  for (int c = 0; c < 272; ++c) acc += sIn[c] * A.Ws0[c * 64 + t];
  hid1[t] = gelu_f(acc);
  __syncthreads();
  acc = A.bs1[t];
  for (int c = 0; c < 64; ++c) acc += hid1[c] * A.Ws1[c * 64 + t];
  hid2[t] = gelu_f(acc);
  __syncthreads();

  for (int o = t; o < STATE; o += 64) {
    float v = A.bs2[o];
    for (int c = 0; c < 64; ++c) v += hid2[c] * A.Ws2[c * STATE + o];
    A.h[i * STATE + o] = sIn[o] + v;
  }
  if (t < 3) {
    float xo = A.x[i * 3 + t];
    A.x[i * 3 + t] = xo + A.dx[i * 3 + t] + A.beta[i] * (A.base[i * 3 + t] - xo);
  }
}

__global__ void copy_out_kernel(const float* x, float* out) {
  int t = blockIdx.x * 256 + threadIdx.x;
  if (t < NN * 3) out[t] = x[t];
}

// ---------------------------------------------------------------------------
// Launch. Input order: Z, xA, xB, bond_mask_A, bond_mask_B, s, then params
// flattened by sorted dict keys: alpha(12), beta(12), embA(4), embB(4),
// emb_table(1), info(4), msg(12), state(12).
// ---------------------------------------------------------------------------
extern "C" void kernel_launch(void* const* d_in, const int* in_sizes, int n_in,
                              void* d_out, int out_size, void* d_ws, size_t ws_size,
                              hipStream_t stream) {
  (void)in_sizes; (void)n_in; (void)out_size; (void)ws_size;
  auto F = [&](int i) -> const float* { return (const float*)d_in[i]; };
  const int*           Z   = (const int*)d_in[0];
  const float*         xA  = F(1);
  const float*         xB  = F(2);
  const unsigned char* bmA = (const unsigned char*)d_in[3];
  const unsigned char* bmB = (const unsigned char*)d_in[4];
  const float*         s   = F(5);

  float* p    = (float*)d_ws;
  float* h    = p; p += NN * STATE;
  float* base = p; p += NN * 3;
  float* x    = p; p += NN * 3;
  float* dx   = p; p += NN * 3;
  float* m    = p; p += NN * 64;
  float* beta = p; p += NN;
  float* Um   = p; p += NN * 64;
  float* Vm   = p; p += NN * 64;
  float* Ua   = p; p += NN * 64;
  float* Va   = p; p += NN * 64;

  EmbArgs ea = { Z, xA, xB, s, F(38),
                 F(39), F(40), F(41), F(42),
                 F(30), F(31), F(32), F(33),
                 F(34), F(35), F(36), F(37),
                 h, base, x };
  node_embed_kernel<<<NN, 64, 0, stream>>>(ea);

  for (int l = 0; l < 2; ++l) {
    const int AM = 6 + l * 6;    // alpha leaves
    const int BM = 18 + l * 6;   // beta leaves
    const int MM = 43 + l * 6;   // msg leaves
    const int SM = 55 + l * 6;   // state leaves

    ProjArgs pa = { h, F(MM + 0), F(MM + 1), F(AM + 0), F(AM + 1),
                    F(BM + 0), F(BM + 1), F(BM + 2), F(BM + 3), F(BM + 4), F(BM + 5),
                    Um, Vm, Ua, Va, beta };
    node_proj_kernel<<<NN, 64, 0, stream>>>(pa);

    PairArgs qa = { x, xA, xB, bmA, bmB, Um, Vm, Ua, Va,
                    F(MM + 0), F(MM + 2), F(MM + 3), F(MM + 4), F(MM + 5),
                    F(AM + 0), F(AM + 2), F(AM + 3), F(AM + 4), F(AM + 5),
                    m, dx };
    pair_kernel<<<NN, 128, 0, stream>>>(qa);

    UpdArgs ua = { m, dx, beta, base,
                   F(SM + 0), F(SM + 1), F(SM + 2), F(SM + 3), F(SM + 4), F(SM + 5),
                   h, x };
    update_kernel<<<NN, 64, 0, stream>>>(ua);
  }

  copy_out_kernel<<<(NN * 3 + 255) / 256, 256, 0, stream>>>(x, (float*)d_out);
}
